// MY_LSTM_58695023067953
// MI455X (gfx1250) — matrix operand order repaired and checked by reading, not compiled, not run on hardware
//
#include <hip/hip_runtime.h>

// ---------------------------------------------------------------------------
// LSTM on MI455X (gfx1250).
//   T=50000, DIMX=DIMH=512, din=1024, 4 gates -> 2048 gate rows.
// Phase 1: Gx = seq @ Wx.T via v_wmma_f32_16x16x32_bf16, register-blocked
//          16x128 per wave; per K-chunk all 8 B fragments are loaded up-front
//          so the 16 b128 loads issue in one clause and waits stagger.
// Phase 2: 16 persistent WGs, 256KB fp32 recurrent weights in LDS per WGP,
//          one device-scope atomic barrier per timestep, double-buffered h.
// ---------------------------------------------------------------------------

#define T_TOTAL   50000
#define CHUNK     2000          // 50000 = 25 * 2000, 2000 % 16 == 0
#define NCHUNK    25
#define G4        2048          // 4 gates * 512
#define NWG       16            // phase-2 workgroups (one per WGP)
#define P2T       128           // phase-2 threads per WG

typedef __attribute__((ext_vector_type(4)))  float   v4f;
typedef __attribute__((ext_vector_type(8)))  float   v8f;
typedef __attribute__((ext_vector_type(8)))  __bf16  v8bf;
typedef __attribute__((ext_vector_type(16))) __bf16  v16bf;

// ---------------------------------------------------------------------------
// Weight prep: Wx (x-part, cols 512..1023) -> bf16 [g=gate*512+j][k],
//              Wh (h-part, cols 0..511)    -> fp32 packed per phase-2 WG:
//              whp[w][k][r] with r = gate*32 + (j%32), w = j/32.
// ---------------------------------------------------------------------------
__global__ void lstm_prep_weights(const float* __restrict__ Wf,
                                  const float* __restrict__ Wi,
                                  const float* __restrict__ Wo,
                                  const float* __restrict__ Wc,
                                  __bf16* __restrict__ wx16,
                                  float*  __restrict__ whp) {
  int idx  = blockIdx.x * 256 + threadIdx.x;   // 0 .. 2048*512-1
  int g    = idx >> 9;
  int k    = idx & 511;
  int gate = g >> 9;
  int j    = g & 511;
  const float* W = (gate == 0) ? Wf : (gate == 1) ? Wi : (gate == 2) ? Wo : Wc;
  wx16[(long)g * 512 + k] = (__bf16)W[(long)j * 1024 + 512 + k];
  int w = j >> 5;
  int r = (gate << 5) + (j & 31);
  whp[((long)w * 512 + k) * 128 + r] = W[(long)j * 1024 + k];
}

// ---------------------------------------------------------------------------
// Phase 1: Gx[sRel][g] = sum_k seq[t][k] * Wgate[j][512+k], bf16 WMMA, f32 acc.
// One wave computes a 16x128 strip: 1 t-tile x 8 g-tiles; K=512 in 16x32.
// Per K-chunk: one A fragment (fp32 load + cvt_pk_bf16) + 8 prefetched B
// fragments feed 8 WMMAs.
// ---------------------------------------------------------------------------
__global__ void lstm_xgemm(const float* __restrict__ seq,
                           const __bf16* __restrict__ wx16,
                           float* __restrict__ gx, int tBase) {
  int wave = threadIdx.x >> 5;
  int lane = threadIdx.x & 31;
  int task = blockIdx.x * 8 + wave;      // 125 t-tiles * 16 g-octets = 2000
  int tt   = task >> 4;                  // 0..124
  int go   = task & 15;                  // g-octet: g-tiles go*8 .. go*8+7
  int mn   = lane & 15;                  // A row (M) / B col (N) for this lane
  int hi   = lane >> 4;                  // lane half

  // A fragment: K = 16*hi + (0..15), contiguous (matches B)
  const float* pa = seq + (long)(tBase + tt * 16 + mn) * 512 + hi * 16;
  // B fragment: K = 16*hi + (0..15), contiguous per g row
  const __bf16* pb = wx16 + (long)(go * 128 + mn) * 512 + hi * 16;

  v8f acc[8] = {};
#pragma unroll
  for (int kc = 0; kc < 16; ++kc) {
    int ko = kc * 32;

    // Issue all B loads for this K-chunk up-front (one big clause; waits
    // stagger because LOADcnt decrements in issue order).
    v16bf Bt[8];
#pragma unroll
    for (int i = 0; i < 8; ++i)
      Bt[i] = *(const v16bf*)(pb + (long)i * 16 * 512 + ko);   // 2x b128 each

    // A: load fp32 and convert (v_cvt_pk_bf16_f32), shared by all 8 WMMAs.
    v4f a0 = *(const v4f*)(pa + ko);
    v4f a1 = *(const v4f*)(pa + ko + 4);
    v4f a2 = *(const v4f*)(pa + ko + 8);
    v4f a3 = *(const v4f*)(pa + ko + 12);
    v16bf A;
#pragma unroll
    for (int i = 0; i < 4; ++i) {
      A[i]      = (__bf16)a0[i];
      A[4 + i]  = (__bf16)a1[i];
      A[8 + i]  = (__bf16)a2[i];
      A[12 + i] = (__bf16)a3[i];
    }

#pragma unroll
    for (int i = 0; i < 8; ++i)
      acc[i] = __builtin_amdgcn_wmma_f32_16x16x32_bf16(
          /*neg_a=*/false, A, /*neg_b=*/false, Bt[i],
          /*c_mod=*/(short)0, acc[i], /*reuse_a=*/false, /*reuse_b=*/false);
  }

  // D: VGPR r -> M = r + 8*hi, N = mn (ISA 7.12.2 C/D layout)
  int sRel = tt * 16 + hi * 8;
#pragma unroll
  for (int i = 0; i < 8; ++i) {
    int g = go * 128 + i * 16 + mn;
#pragma unroll
    for (int r = 0; r < 8; ++r)
      gx[(long)(sRel + r) * G4 + g] = acc[i][r];
  }
}

// ---------------------------------------------------------------------------
// Phase 2: persistent recurrence. 16 WGs x 128 threads.
// Thread tid: gate = tid/32, j = blockIdx*32 + tid%32 (one gate row each).
// Dynamic LDS: wT[512][128] fp32 (256 KB) -> conflict-free ds loads.
// ---------------------------------------------------------------------------
__global__ void lstm_recur(const float* __restrict__ gx,
                           const float* __restrict__ whp,
                           float* hbuf, float* cstate, unsigned int* ctr,
                           const float* __restrict__ bfv,
                           const float* __restrict__ biv,
                           const float* __restrict__ bov,
                           const float* __restrict__ bcv,
                           float* __restrict__ out, int tBase, int steps) {
  extern __shared__ float wT[];          // 512*128 floats = 256 KB
  __shared__ float h_s[512];
  __shared__ float gate_s[128];

  int tid  = threadIdx.x;
  int w    = blockIdx.x;                 // 0..15
  int gate = tid >> 5;
  int j    = (w << 5) + (tid & 31);

  // Stage this WG's recurrent weights into LDS (vectorized, coalesced).
  const float4* wsrc4 = (const float4*)(whp + (long)w * 512 * 128);
  for (int i = tid; i < 512 * 128 / 4; i += P2T)
    ((float4*)wT)[i] = wsrc4[i];

  const float* bsel = (gate == 0) ? bfv : (gate == 1) ? biv : (gate == 2) ? bov : bcv;
  float bias = bsel[j];
  float c = 0.f;
  if (tid < 32) c = cstate[(w << 5) + tid];
  __syncthreads();

  for (int s = 0; s < steps; ++s) {
    int t = tBase + s;

    // Wait until all 16 WGs published h_t (global monotonic counter).
    if (tid == 0) {
      unsigned int need = 16u * (unsigned int)t;
      while (__hip_atomic_load(ctr, __ATOMIC_ACQUIRE, __HIP_MEMORY_SCOPE_AGENT) < need)
        __builtin_amdgcn_s_sleep(1);
    }
    __syncthreads();

    // Broadcast h_t into LDS (double-buffered in global).
    const float* hsrc = hbuf + (t & 1) * 512;
#pragma unroll
    for (int i = 0; i < 4; ++i) h_s[tid + i * P2T] = hsrc[tid + i * P2T];
    __syncthreads();

    // acc = Gx + bias + h_t . Wh[row]   (4 independent FMA chains)
    float a0 = gx[(long)s * G4 + (gate << 9) + j] + bias;
    float a1 = 0.f, a2 = 0.f, a3 = 0.f;
    const float* wr = wT + tid;
    for (int k = 0; k < 512; k += 4) {
      a0 += h_s[k]     * wr[(k)     * 128];
      a1 += h_s[k + 1] * wr[(k + 1) * 128];
      a2 += h_s[k + 2] * wr[(k + 2) * 128];
      a3 += h_s[k + 3] * wr[(k + 3) * 128];
    }
    gate_s[tid] = (a0 + a1) + (a2 + a3);
    __syncthreads();

    if (tid < 32) {
      float fg = 1.f / (1.f + expf(-gate_s[tid]));
      float ig = 1.f / (1.f + expf(-gate_s[32 + tid]));
      float og = 1.f / (1.f + expf(-gate_s[64 + tid]));
      float cc = tanhf(gate_s[96 + tid]);
      c = fg * c + ig * cc;
      float h = og * tanhf(c);
      int jj = (w << 5) + tid;
      hbuf[((t + 1) & 1) * 512 + jj] = h;
      out[(long)(t + 1) * 512 + jj]  = h;
      __threadfence();                   // make h_{t+1} device-visible
    }
    __syncthreads();
    if (tid == 0)
      __hip_atomic_fetch_add(ctr, 1u, __ATOMIC_RELEASE, __HIP_MEMORY_SCOPE_AGENT);
  }

  if (tid < 32) cstate[(w << 5) + tid] = c;   // persist cell state across chunks
}

// ---------------------------------------------------------------------------
// Workspace layout (bytes):
//   [0]        ctr (u32), zeroed per launch
//   [256]      hbuf: 2 x 512 f32 (double-buffered h), zeroed per launch
//   [4352]     cstate: 512 f32, zeroed per launch
//   [8192]     wx16: 2048*512 bf16  (2 MB)
//   [2105344]  whp : 2048*512 f32   (4 MB)
//   [6299648]  gx  : CHUNK*2048 f32 (16 MB, stays L2-resident)
// total ~22.7 MB
// ---------------------------------------------------------------------------
extern "C" void kernel_launch(void* const* d_in, const int* in_sizes, int n_in,
                              void* d_out, int out_size, void* d_ws, size_t ws_size,
                              hipStream_t stream) {
  (void)in_sizes; (void)n_in; (void)out_size; (void)ws_size;
  const float* seq = (const float*)d_in[0];
  const float* Wf  = (const float*)d_in[1];
  const float* bf_ = (const float*)d_in[2];
  const float* Wi  = (const float*)d_in[3];
  const float* bi_ = (const float*)d_in[4];
  const float* Wo  = (const float*)d_in[5];
  const float* bo_ = (const float*)d_in[6];
  const float* Wc  = (const float*)d_in[7];
  const float* bc_ = (const float*)d_in[8];
  float* out = (float*)d_out;

  char* ws = (char*)d_ws;
  unsigned int* ctr = (unsigned int*)(ws + 0);
  float*  hbuf = (float*)(ws + 256);
  float*  cst  = (float*)(ws + 4352);
  __bf16* wx16 = (__bf16*)(ws + 8192);
  float*  whp  = (float*)(ws + 2105344);
  float*  gx   = (float*)(ws + 6299648);

  // Zero sync/state region and the h0 row of the output (graph-replay safe).
  hipMemsetAsync(ws, 0, 6400, stream);
  hipMemsetAsync(out, 0, 512 * sizeof(float), stream);

  // One-time weight repack (bf16 x-part, LDS-layout h-part).
  lstm_prep_weights<<<(G4 * 512) / 256, 256, 0, stream>>>(Wf, Wi, Wo, Wc, wx16, whp);

  // Allow 256 KB dynamic LDS for the persistent recurrence kernel.
  hipFuncSetAttribute((const void*)lstm_recur,
                      hipFuncAttributeMaxDynamicSharedMemorySize, 512 * 128 * 4);

  for (int cix = 0; cix < NCHUNK; ++cix) {
    int tBase = cix * CHUNK;
    // 125 t-tiles * 16 g-octets = 2000 wave-tasks, 8 waves per 256-thread block.
    lstm_xgemm<<<250, 256, 0, stream>>>(seq, wx16, gx, tBase);
    lstm_recur<<<NWG, P2T, 512 * 128 * 4, stream>>>(gx, whp, hbuf, cst, ctr,
                                                    bf_, bi_, bo_, bc_,
                                                    out, tBase, CHUNK);
  }
}
